// MixtureOfExperts_43868795961385
// MI455X (gfx1250) — matrix operand order repaired and checked by reading, not compiled, not run on hardware
//
#include <hip/hip_runtime.h>
#include <hip/hip_bf16.h>
#include <math.h>

// Problem constants (match reference)
#define B_TOK 8192
#define DIM   2048
#define NEXP  8

// GEMM tiling
#define BM 128
#define BN 128
#define KT 32     // K-slice (bf16 WMMA depth)
#define LSTRB 40  // padded LDS row stride in bf16 (80B: 16B-aligned, conflict-free)

typedef float v8f  __attribute__((ext_vector_type(8)));
typedef __bf16 v16bf __attribute__((ext_vector_type(16)));
typedef __bf16 v8bf  __attribute__((ext_vector_type(8)));

// ---------------------------------------------------------------------------
// Kernel 1: zero output buffer + routing counters
// ---------------------------------------------------------------------------
__global__ void moe_zero_kernel(float* __restrict__ out, int* __restrict__ cnt, int n) {
    int tid = blockIdx.x * blockDim.x + threadIdx.x;
    int n4 = n >> 2;
    float4* out4 = (float4*)out;
    for (int i = tid; i < n4; i += gridDim.x * blockDim.x) {
        out4[i] = make_float4(0.f, 0.f, 0.f, 0.f);
    }
    if (tid < NEXP) cnt[tid] = 0;
}

// ---------------------------------------------------------------------------
// Kernel 2: gating — softmax over 8 experts, top-2, append to expert lists
// ---------------------------------------------------------------------------
__global__ void moe_gating_kernel(const float* __restrict__ x,
                                  const float* __restrict__ Wg,
                                  const float* __restrict__ bg,
                                  int* __restrict__ cnt,
                                  int* __restrict__ tokList,
                                  float* __restrict__ wgtList) {
    int b = blockIdx.x * blockDim.x + threadIdx.x;
    if (b >= B_TOK) return;

    const float* xrow = x + (size_t)b * DIM;
    float acc[NEXP];
#pragma unroll
    for (int e = 0; e < NEXP; ++e) acc[e] = 0.f;

    for (int d = 0; d < DIM; ++d) {
        float xv = xrow[d];
#pragma unroll
        for (int e = 0; e < NEXP; ++e) {
            acc[e] = fmaf(xv, Wg[e * DIM + d], acc[e]);
        }
    }
#pragma unroll
    for (int e = 0; e < NEXP; ++e) acc[e] += bg[e];

    float m = acc[0];
#pragma unroll
    for (int e = 1; e < NEXP; ++e) m = fmaxf(m, acc[e]);
    float s = 0.f;
    float p[NEXP];
#pragma unroll
    for (int e = 0; e < NEXP; ++e) { p[e] = __expf(acc[e] - m); s += p[e]; }
    float inv = 1.f / s;
#pragma unroll
    for (int e = 0; e < NEXP; ++e) p[e] *= inv;

    // top-2 (ties -> lowest index, matching lax.top_k)
    int e0 = 0;
#pragma unroll
    for (int e = 1; e < NEXP; ++e) if (p[e] > p[e0]) e0 = e;
    int e1 = (e0 == 0) ? 1 : 0;
#pragma unroll
    for (int e = 0; e < NEXP; ++e) {
        if (e != e0 && p[e] > p[e1]) e1 = e;
    }

    int pos0 = atomicAdd(&cnt[e0], 1);
    tokList[e0 * B_TOK + pos0] = b;
    wgtList[e0 * B_TOK + pos0] = p[e0];

    int pos1 = atomicAdd(&cnt[e1], 1);
    tokList[e1 * B_TOK + pos1] = b;
    wgtList[e1 * B_TOK + pos1] = p[e1];
}

// ---------------------------------------------------------------------------
// helpers
// ---------------------------------------------------------------------------
__device__ __forceinline__ unsigned pk2(__bf16 a, __bf16 b) {
    unsigned short ua = __builtin_bit_cast(unsigned short, a);
    unsigned short ub = __builtin_bit_cast(unsigned short, b);
    return (unsigned)ua | ((unsigned)ub << 16);
}

// Load a 16-contiguous-bf16 fragment (two 16B ds_load_b128) from a plane row.
// Both A (pre-swizzled) and B (same swizzle) planes put each lane-half's 16
// fragment values contiguously at element offset khalf*16.
__device__ __forceinline__ v16bf load_frag32(const __bf16* plane, int row, int khalf) {
    const v8bf* p = (const v8bf*)(plane + row * LSTRB + khalf * 16);
    v8bf lo = p[0];
    v8bf hi = p[1];
    return __builtin_shufflevector(lo, hi, 0, 1, 2, 3, 4, 5, 6, 7,
                                   8, 9, 10, 11, 12, 13, 14, 15);
}

// ---------------------------------------------------------------------------
// Kernel 3: routed expert GEMM, fp32 emulated via bf16x3 WMMA.
// grid = (DIM/BN, B_TOK/BM, NEXP); 256 threads = 8 waves (wave32).
// Wave grid 4(M) x 2(N); each wave computes 32x64 via 2x4 fragments,
// 3 bf16 WMMAs per fragment per K=32 slice (hi*hi + hi*lo + lo*hi).
// Double-buffered LDS (one barrier per slice), fragment-ready swizzled layout.
//
// A fragment layout (ISA 7.12.2, 16-bit A 16x32): lane half 0 consumes
// K {0..7,16..23}, half 1 consumes K {8..15,24..31} -> row stored permuted as
// [K0-7 | K16-23 | K8-15 | K24-31]. B (32x16) rows are stored in the same permuted
// order, so B half h gets the same K values as A half h.
// ---------------------------------------------------------------------------
__global__ __launch_bounds__(256)
void moe_gemm_kernel(const float* __restrict__ x,
                     const float* __restrict__ W,     // [E, D, D] (out, in)
                     const float* __restrict__ bias,  // [E, D]
                     const int* __restrict__ cnt,
                     const int* __restrict__ tokList,
                     const float* __restrict__ wgtList,
                     float* __restrict__ out) {
    __shared__ __bf16 XsH[2 * BM * LSTRB];
    __shared__ __bf16 XsL[2 * BM * LSTRB];
    __shared__ __bf16 WsH[2 * BN * LSTRB];
    __shared__ __bf16 WsL[2 * BN * LSTRB];
    __shared__ int    tokS[BM];
    __shared__ float  wgtS[BM];

    const int e  = blockIdx.z;
    const int m0 = blockIdx.y * BM;
    const int n0 = blockIdx.x * BN;

    const int mcount = cnt[e];
    if (m0 >= mcount) return;

    const int tid    = threadIdx.x;
    const int lane   = tid & 31;
    const int waveId = tid >> 5;
    const int waveM  = waveId >> 1;    // 0..3
    const int waveN  = waveId & 1;     // 0..1
    const int mrow   = lane & 15;
    const int khalf  = lane >> 4;      // 0 or 1

    if (tid < BM) {
        int gm = m0 + tid;
        if (gm < mcount) {
            tokS[tid] = tokList[e * B_TOK + gm];
            wgtS[tid] = wgtList[e * B_TOK + gm];
        } else {
            tokS[tid] = 0;
            wgtS[tid] = 0.f;
        }
    }
    __syncthreads();

    v8f acc[2][4] = {};

    // tile fill: 2 threads per row, 16 contiguous floats each (row has KT=32)
    const int lr = tid >> 1;           // 0..127
    const int lc = (tid & 1) * 16;     // 0 or 16
    const int xtok = tokS[lr];
    const float* xbase = x + (size_t)xtok * DIM + lc;
    const float* wbase = W + ((size_t)e * DIM + (n0 + lr)) * (size_t)DIM + lc;

    // store positions (element offsets within a row)
    const int posA0 = (tid & 1) * 8;        // k chunk0 -> swizzled slot
    const int posA1 = 16 + (tid & 1) * 8;   // k chunk1 -> swizzled slot
    const int posB0 = (tid & 1) * 8;        // same swizzle as A
    const int posB1 = 16 + (tid & 1) * 8;

    // slice registers (packed bf16 pairs, 16B each)
    uint4 xh0, xh1, xl0, xl1, wh0, wh1, wl0, wl1;

    // ---- load + fp32->bf16(hi/lo) split + pack ----
    auto load_convert = [&](int kt) {
        float xv[16], wv[16];
#pragma unroll
        for (int q = 0; q < 4; ++q) {
            float4 a = *(const float4*)(xbase + kt + 4 * q);
            float4 b = *(const float4*)(wbase + kt + 4 * q);
            xv[4 * q + 0] = a.x; xv[4 * q + 1] = a.y; xv[4 * q + 2] = a.z; xv[4 * q + 3] = a.w;
            wv[4 * q + 0] = b.x; wv[4 * q + 1] = b.y; wv[4 * q + 2] = b.z; wv[4 * q + 3] = b.w;
        }
        // prefetch the slice after next
        __builtin_prefetch(xbase + kt + KT, 0, 1);
        __builtin_prefetch(wbase + kt + KT, 0, 1);

        __bf16 xh[16], xl[16], wh[16], wl[16];
#pragma unroll
        for (int q = 0; q < 16; ++q) {
            xh[q] = (__bf16)xv[q];
            xl[q] = (__bf16)(xv[q] - (float)xh[q]);
            wh[q] = (__bf16)wv[q];
            wl[q] = (__bf16)(wv[q] - (float)wh[q]);
        }
        xh0 = make_uint4(pk2(xh[0], xh[1]), pk2(xh[2], xh[3]), pk2(xh[4], xh[5]), pk2(xh[6], xh[7]));
        xh1 = make_uint4(pk2(xh[8], xh[9]), pk2(xh[10], xh[11]), pk2(xh[12], xh[13]), pk2(xh[14], xh[15]));
        xl0 = make_uint4(pk2(xl[0], xl[1]), pk2(xl[2], xl[3]), pk2(xl[4], xl[5]), pk2(xl[6], xl[7]));
        xl1 = make_uint4(pk2(xl[8], xl[9]), pk2(xl[10], xl[11]), pk2(xl[12], xl[13]), pk2(xl[14], xl[15]));
        wh0 = make_uint4(pk2(wh[0], wh[1]), pk2(wh[2], wh[3]), pk2(wh[4], wh[5]), pk2(wh[6], wh[7]));
        wh1 = make_uint4(pk2(wh[8], wh[9]), pk2(wh[10], wh[11]), pk2(wh[12], wh[13]), pk2(wh[14], wh[15]));
        wl0 = make_uint4(pk2(wl[0], wl[1]), pk2(wl[2], wl[3]), pk2(wl[4], wl[5]), pk2(wl[6], wl[7]));
        wl1 = make_uint4(pk2(wl[8], wl[9]), pk2(wl[10], wl[11]), pk2(wl[12], wl[13]), pk2(wl[14], wl[15]));
    };

    load_convert(0);

    int buf = 0;
    for (int kt = 0; kt < DIM; kt += KT) {
        // ---- store slice into current LDS buffer (8x ds_store_b128) ----
        {
            const int bofs = buf * BM * LSTRB;
            const int rofs = bofs + lr * LSTRB;
            *(uint4*)(XsH + rofs + posA0) = xh0;
            *(uint4*)(XsH + rofs + posA1) = xh1;
            *(uint4*)(XsL + rofs + posA0) = xl0;
            *(uint4*)(XsL + rofs + posA1) = xl1;
            *(uint4*)(WsH + rofs + posB0) = wh0;
            *(uint4*)(WsH + rofs + posB1) = wh1;
            *(uint4*)(WsL + rofs + posB0) = wl0;
            *(uint4*)(WsL + rofs + posB1) = wl1;
        }
        __syncthreads();

        // issue next slice's global loads before consuming this one
        if (kt + KT < DIM) load_convert(kt + KT);

        // ---- consume slice: fragments + 24 bf16 WMMAs ----
        const __bf16* pXH = XsH + buf * BM * LSTRB;
        const __bf16* pXL = XsL + buf * BM * LSTRB;
        const __bf16* pWH = WsH + buf * BN * LSTRB;
        const __bf16* pWL = WsL + buf * BN * LSTRB;

        v16bf aH[2], aL[2], bH[4], bL[4];
#pragma unroll
        for (int i = 0; i < 2; ++i) {
            int row = waveM * 32 + i * 16 + mrow;
            aH[i] = load_frag32(pXH, row, khalf);
            aL[i] = load_frag32(pXL, row, khalf);
        }
#pragma unroll
        for (int j = 0; j < 4; ++j) {
            int row = waveN * 64 + j * 16 + mrow;
            bH[j] = load_frag32(pWH, row, khalf);
            bL[j] = load_frag32(pWL, row, khalf);
        }

#pragma unroll
        for (int i = 0; i < 2; ++i) {
#pragma unroll
            for (int j = 0; j < 4; ++j) {
                acc[i][j] = __builtin_amdgcn_wmma_f32_16x16x32_bf16(
                    false, aH[i], false, bH[j], (short)0, acc[i][j], false, false);
                acc[i][j] = __builtin_amdgcn_wmma_f32_16x16x32_bf16(
                    false, aH[i], false, bL[j], (short)0, acc[i][j], false, false);
                acc[i][j] = __builtin_amdgcn_wmma_f32_16x16x32_bf16(
                    false, aL[i], false, bH[j], (short)0, acc[i][j], false, false);
            }
        }
        buf ^= 1;
    }

    // epilogue: +bias, ReLU, scale by routing prob, scatter-accumulate
    const int halfM = (lane >> 4) * 8;   // C layout: lanes 16-31 hold M = v + 8
    const int ncol  = lane & 15;
#pragma unroll
    for (int i = 0; i < 2; ++i) {
#pragma unroll
        for (int j = 0; j < 4; ++j) {
            int n_local  = waveN * 64 + j * 16 + ncol;
            int n_global = n0 + n_local;
            float bn = bias[e * DIM + n_global];
#pragma unroll
            for (int v = 0; v < 8; ++v) {
                int m_local = waveM * 32 + i * 16 + v + halfM;
                int gm = m0 + m_local;
                if (gm < mcount) {
                    float val = fmaxf(acc[i][j][v] + bn, 0.f) * wgtS[m_local];
                    atomicAdd(&out[(size_t)tokS[m_local] * DIM + n_global], val);
                }
            }
        }
    }
}

// ---------------------------------------------------------------------------
// Host-side launch
// ---------------------------------------------------------------------------
extern "C" void kernel_launch(void* const* d_in, const int* in_sizes, int n_in,
                              void* d_out, int out_size, void* d_ws, size_t ws_size,
                              hipStream_t stream) {
    const float* x  = (const float*)d_in[0];   // [B, D]
    const float* W  = (const float*)d_in[1];   // [E, D, D]
    const float* bb = (const float*)d_in[2];   // [E, D]
    const float* Wg = (const float*)d_in[3];   // [E, D]
    const float* bg = (const float*)d_in[4];   // [E]
    float* out = (float*)d_out;

    char* ws = (char*)d_ws;
    int*   cnt     = (int*)ws;                                  // 8 ints (256B slot)
    int*   tokList = (int*)(ws + 256);                          // [E][B]
    float* wgtList = (float*)(ws + 256 + NEXP * B_TOK * 4);     // [E][B]

    {
        int n = B_TOK * DIM;
        int blocks = (n / 4 + 255) / 256;
        if (blocks > 4096) blocks = 4096;
        moe_zero_kernel<<<blocks, 256, 0, stream>>>(out, cnt, n);
    }
    {
        dim3 grid((B_TOK + 255) / 256);
        moe_gating_kernel<<<grid, 256, 0, stream>>>(x, Wg, bg, cnt, tokList, wgtList);
    }
    {
        dim3 grid(DIM / BN, B_TOK / BM, NEXP);
        moe_gemm_kernel<<<grid, 256, 0, stream>>>(x, W, bb, cnt, tokList, wgtList, out);
    }
}
